// ModuleNet_81106162418251
// MI455X (gfx1250) — compile-verified, matrix operand order checked by reading, not run on hardware
//
#include <hip/hip_runtime.h>
#include <hip/hip_bf16.h>

// ---------------------------------------------------------------------------
// ModuleNet on MI455X (gfx1250): bucket-by-module gather-GEMM, all matmuls on
// v_wmma_f32_16x16x32_bf16 (fp32 accumulate), weights resident in LDS.
// Round 3: __launch_bounds__(256,1) + explicit software pipelining of B/A
// fragment ds_load_b128s across WMMA tiles (kills the ds_wait-0 serialization
// seen from tile 2 onward). LDS rows stay +16B padded for 64-bank spread.
// ---------------------------------------------------------------------------

typedef __attribute__((ext_vector_type(16))) __bf16 v16bf;
typedef __attribute__((ext_vector_type(8)))  __bf16 v8bf;
typedef __attribute__((ext_vector_type(8)))  float  v8f;

#define NB     4096        // batch
#define NE     128         // embedding dim
#define NM     8           // modules
#define NSTEPS 3
#define NH     256         // head hidden
#define NC     16          // classes
#define ROWS   64          // samples per block in step/head kernels

#define LDK128 136         // padded LDS row stride for K=128 tiles (272B = 68 dwords)
#define LDK256 264         // padded LDS row stride for K=256 tiles (528B = 132 dwords)

// ---- workspace layout (bytes) ----
static constexpr size_t OFF_WIN = 0;                         // M*E*E bf16 (W_in^T)
static constexpr size_t OFF_WBI = 262144;                    // M*E*E bf16 (W_bias^T)
static constexpr size_t OFF_WF  = 524288;                    // M*E*2E bf16 (W_f^T)
static constexpr size_t OFF_W1  = 1048576;                   // H*E bf16 (W1^T)
static constexpr size_t OFF_W2  = 1114112;                   // C*H bf16 (W2^T)
static constexpr size_t OFF_CNT = 1122304;                   // 3*8 int counts
static constexpr size_t OFF_LST = 1122432;                   // 3*8*B int lists
static constexpr size_t OFF_X0  = 1515648;                   // B*E fp32 state ping
static constexpr size_t OFF_X1  = 3612800;                   // B*E fp32 state pong

static constexpr unsigned STEP_LDS =
    (2 * 128 * LDK128 + 128 * LDK256 + 2 * 64 * LDK128 + 64 * LDK256) * 2 + 256; // 206080 B
static constexpr unsigned HEAD_LDS =
    (256 * LDK128 + 16 * LDK256 + 64 * LDK128 + 64 * LDK256) * 2;                // 129280 B

// Assemble a 16-bit WMMA A/B fragment for one 32-wide K chunk from an LDS row.
// Per ISA 16-bit A-layout: lanes 0-15 take K [kb..kb+7] and [kb+16..kb+23],
// lanes 16-31 (kg=1) take K [kb+8..kb+15] and [kb+24..kb+31].
__device__ __forceinline__ v16bf load_frag(const __bf16* row, int kb, int kg) {
  const v8bf lo = *reinterpret_cast<const v8bf*>(row + kb + kg * 8);
  const v8bf hi = *reinterpret_cast<const v8bf*>(row + kb + 16 + kg * 8);
  return __builtin_shufflevector(lo, hi, 0, 1, 2, 3, 4, 5, 6, 7,
                                          8, 9, 10, 11, 12, 13, 14, 15);
}

__device__ __forceinline__ v8f wmma_bf16(v16bf a, v16bf b, v8f c) {
  return __builtin_amdgcn_wmma_f32_16x16x32_bf16(false, a, false, b, (short)0, c,
                                                 false, false);
}

// Copy `nrows` rows of `ncols` (multiple of 8) bf16 from dense global into LDS
// with padded row stride `ldst`. b128 on both sides.
__device__ __forceinline__ void copy_w(const __bf16* __restrict__ g, __bf16* l,
                                       int nrows, int ncols, int ldst, int tid) {
  const int segs = ncols >> 3;
  const v8bf* gv = (const v8bf*)g;
  for (int i = tid; i < nrows * segs; i += 256) {
    int r = i / segs, s = i - r * segs;
    *reinterpret_cast<v8bf*>(l + r * ldst + s * 8) = gv[i];
  }
}

// ---------------------------------------------------------------------------
// Preprocess: transpose all weights to N-major bf16 so B-fragments are
// K-contiguous in LDS.
// ---------------------------------------------------------------------------
__global__ void prep_kernel(const float* __restrict__ Win, const float* __restrict__ Wbi,
                            const float* __restrict__ Wf,  const float* __restrict__ W1,
                            const float* __restrict__ W2,
                            __bf16* __restrict__ Wint, __bf16* __restrict__ Wbit,
                            __bf16* __restrict__ Wft,  __bf16* __restrict__ W1t,
                            __bf16* __restrict__ W2t) {
  int i = blockIdx.x * blockDim.x + threadIdx.x;
  const int S1 = NM * NE * NE;       // 131072 (Win & Wbi together)
  const int S3 = NM * 2 * NE * NE;   // 262144
  const int S4 = NE * NH;            // 32768
  const int S5 = NH * NC;            // 4096
  if (i < S1) {                      // Wt[m][o][e] = W[m][e][o]
    int m = i >> 14, o = (i >> 7) & 127, e = i & 127;
    int src = (m << 14) + (e << 7) + o;
    Wint[i] = (__bf16)Win[src];
    Wbit[i] = (__bf16)Wbi[src];
    return;
  }
  i -= S1;
  if (i < S3) {                      // Wft[m][o][f] = W_f[m][f][o]
    int m = i >> 15, o = (i >> 8) & 127, f = i & 255;
    Wft[i] = (__bf16)Wf[(m << 15) + (f << 7) + o];
    return;
  }
  i -= S3;
  if (i < S4) {                      // W1t[h][e] = W1[e][h]
    int h = i >> 7, e = i & 127;
    W1t[i] = (__bf16)W1[(e << 8) + h];
    return;
  }
  i -= S4;
  if (i < S5) {                      // W2t[c][h] = W2[h][c]
    int c = i >> 8, h = i & 255;
    W2t[i] = (__bf16)W2[(h << 4) + c];
  }
}

__global__ void zero_counts(int* counts) {
  if (threadIdx.x < NSTEPS * NM) counts[threadIdx.x] = 0;
}

__global__ void bucketize(const int* __restrict__ module_ids, int* __restrict__ counts,
                          int* __restrict__ lists) {
  int t = blockIdx.x * blockDim.x + threadIdx.x;
  if (t >= NSTEPS * NB) return;
  int s = t / NB, b = t - s * NB;
  int m = module_ids[b * NSTEPS + s];
  int slot = atomicAdd(&counts[s * NM + m], 1);
  lists[(s * NM + m) * NB + slot] = b;
}

__global__ void gather_x0(const int* __restrict__ entity_ids,
                          const float* __restrict__ emb, float* __restrict__ X0) {
  int t = blockIdx.x * blockDim.x + threadIdx.x;   // B*E threads exactly
  int b = t >> 7, e = t & 127;
  int eid = entity_ids[b * (NSTEPS + 1)];
  X0[t] = emb[(size_t)eid * NE + e];
}

// ---------------------------------------------------------------------------
// One recurrent step for one module bucket. Block: 256 threads (8 wave32),
// 64 samples. LDS: W_in^T + W_bias^T + W_f^T + A tiles + h (all row-padded).
// ---------------------------------------------------------------------------
__global__ __launch_bounds__(256, 1) void step_kernel(
    const float* __restrict__ X_in, float* __restrict__ X_out,
    const float* __restrict__ embedding, const int* __restrict__ entity_ids,
    const float* __restrict__ b_in, const float* __restrict__ b_bias,
    const float* __restrict__ b_f,
    const __bf16* __restrict__ Wint, const __bf16* __restrict__ Wbit,
    const __bf16* __restrict__ Wft,
    const int* __restrict__ counts, const int* __restrict__ lists, int step) {
  const int m   = blockIdx.y;
  const int cnt = counts[step * NM + m];
  const int row0 = blockIdx.x * ROWS;
  if (row0 >= cnt) return;

  extern __shared__ __align__(16) char smem[];
  __bf16* sWin = (__bf16*)smem;              // 128 x LDK128
  __bf16* sWbi = sWin + 128 * LDK128;        // 128 x LDK128
  __bf16* sWf  = sWbi + 128 * LDK128;        // 128 x LDK256
  __bf16* sX   = sWf + 128 * LDK256;         // 64 x LDK128
  __bf16* sB   = sX + 64 * LDK128;           // 64 x LDK128
  __bf16* sH   = sB + 64 * LDK128;           // 64 x LDK256
  int*    sIdx = (int*)(sH + 64 * LDK256);   // 64

  const int tid = threadIdx.x;

  // cooperative weight load (b128 global -> b128 LDS, stride-padded rows)
  copy_w(Wint + m * 16384, sWin, 128, 128, LDK128, tid);
  copy_w(Wbit + m * 16384, sWbi, 128, 128, LDK128, tid);
  copy_w(Wft  + m * 32768, sWf,  128, 256, LDK256, tid);
  for (int r = tid; r < ROWS; r += 256)
    sIdx[r] = (row0 + r < cnt) ? lists[(step * NM + m) * NB + row0 + r] : -1;
  __syncthreads();

  // gather activation + embedding rows, fp32 -> bf16
  {
    int r = tid >> 2, c0 = (tid & 3) << 5;
    int si = sIdx[r];
    if (si >= 0) {
      const float* xr = X_in + si * NE;
      int eid = entity_ids[si * (NSTEPS + 1) + step + 1];
      const float* br = embedding + (size_t)eid * NE;
#pragma unroll
      for (int c = 0; c < 32; ++c) {
        sX[r * LDK128 + c0 + c] = (__bf16)xr[c0 + c];
        sB[r * LDK128 + c0 + c] = (__bf16)br[c0 + c];
      }
    } else {
#pragma unroll
      for (int c = 0; c < 32; ++c) {
        sX[r * LDK128 + c0 + c] = (__bf16)0.f;
        sB[r * LDK128 + c0 + c] = (__bf16)0.f;
      }
    }
  }
  __syncthreads();

  const int lane = tid & 31, wave = tid >> 5;
  const int lrow = lane & 15, kg = lane >> 4;

  // ---- phase 1: h_in = relu(x*W_in + b_in), h_bi = relu(bias*W_bias + b_bias)
  // wave: half = wave>>2 selects path, rt = wave&3 selects 16-row strip.
  // Software pipeline: B fragments of tile nt+1 are issued before the WMMAs
  // of tile nt so ds latency overlaps the matrix pipe.
  {
    const int half = wave >> 2, rt = wave & 3;
    const __bf16* A    = half ? sB : sX;
    const __bf16* W    = half ? sWbi : sWin;
    const float*  bias = (half ? b_bias : b_in) + m * NE;
    const __bf16* Arow = A + (rt * 16 + lrow) * LDK128;
    v16bf af[4], bf[4], bn[4];
#pragma unroll
    for (int kc = 0; kc < 4; ++kc) af[kc] = load_frag(Arow, kc * 32, kg);
    const __bf16* Brow0 = W + lrow * LDK128;
#pragma unroll
    for (int kc = 0; kc < 4; ++kc) bf[kc] = load_frag(Brow0, kc * 32, kg);
#pragma unroll
    for (int nt = 0; nt < 8; ++nt) {
      if (nt < 7) {
        const __bf16* Bn = W + ((nt + 1) * 16 + lrow) * LDK128;
#pragma unroll
        for (int kc = 0; kc < 4; ++kc) bn[kc] = load_frag(Bn, kc * 32, kg);
      }
      v8f c = {};
#pragma unroll
      for (int kc = 0; kc < 4; ++kc) c = wmma_bf16(af[kc], bf[kc], c);
      const float bv = bias[nt * 16 + lrow];
      const int ccol = half * NE + nt * 16 + lrow;
      const int crow0 = rt * 16 + 8 * kg;
#pragma unroll
      for (int i = 0; i < 8; ++i) {
        float v = c[i] + bv;
        sH[(crow0 + i) * LDK256 + ccol] = (__bf16)(v > 0.f ? v : 0.f);
      }
      if (nt < 7) {
#pragma unroll
        for (int kc = 0; kc < 4; ++kc) bf[kc] = bn[kc];
      }
    }
  }
  __syncthreads();

  // ---- phase 2: x_new = tanh(h * W_f + b_f), h is 64x256
  // 1-chunk lookahead on B fragments across the whole (tile, chunk) sequence.
  {
    const int rt = wave >> 1, colg = wave & 1;
    const __bf16* Hrow = sH + (rt * 16 + lrow) * LDK256;
    v16bf ah[8];
#pragma unroll
    for (int kc = 0; kc < 8; ++kc) ah[kc] = load_frag(Hrow, kc * 32, kg);
    const int ntb = colg * 4;
    v16bf bcur = load_frag(sWf + (ntb * 16 + lrow) * LDK256, 0, kg);
#pragma unroll
    for (int t = 0; t < 4; ++t) {
      const int nt = ntb + t;
      const __bf16* Brow  = sWf + (nt * 16 + lrow) * LDK256;
      v8f c = {};
#pragma unroll
      for (int kc = 0; kc < 8; ++kc) {
        v16bf bnxt = bcur;
        if (kc < 7)
          bnxt = load_frag(Brow, (kc + 1) * 32, kg);
        else if (t < 3)
          bnxt = load_frag(sWf + ((nt + 1) * 16 + lrow) * LDK256, 0, kg);
        c = wmma_bf16(ah[kc], bcur, c);
        bcur = bnxt;
      }
      const float bv = b_f[m * NE + nt * 16 + lrow];
      const int ccol = nt * 16 + lrow;
      const int crow0 = rt * 16 + 8 * kg;
#pragma unroll
      for (int i = 0; i < 8; ++i) {
        int si = sIdx[crow0 + i];
        if (si >= 0) X_out[si * NE + ccol] = tanhf(c[i] + bv);
      }
    }
  }
}

// ---------------------------------------------------------------------------
// Head: out = relu(x*W1 + b1) * W2 + b2. 64 rows per block.
// ---------------------------------------------------------------------------
__global__ __launch_bounds__(256, 1) void head_kernel(
    const float* __restrict__ X, const __bf16* __restrict__ W1t,
    const __bf16* __restrict__ W2t, const float* __restrict__ b1,
    const float* __restrict__ b2, float* __restrict__ out) {
  extern __shared__ __align__(16) char smem[];
  __bf16* sW1 = (__bf16*)smem;             // 256 x LDK128
  __bf16* sW2 = sW1 + 256 * LDK128;        // 16 x LDK256
  __bf16* sX  = sW2 + 16 * LDK256;         // 64 x LDK128
  __bf16* sH  = sX + 64 * LDK128;          // 64 x LDK256

  const int tid = threadIdx.x;
  const int row0 = blockIdx.x * ROWS;

  copy_w(W1t, sW1, 256, 128, LDK128, tid);
  copy_w(W2t, sW2, 16, 256, LDK256, tid);
  {
    int r = tid >> 2, c0 = (tid & 3) << 5;
    const float* xr = X + (row0 + r) * NE;
#pragma unroll
    for (int c = 0; c < 32; ++c) sX[r * LDK128 + c0 + c] = (__bf16)xr[c0 + c];
  }
  __syncthreads();

  const int lane = tid & 31, wave = tid >> 5;
  const int lrow = lane & 15, kg = lane >> 4;

  // GEMM1: 64x256 over K=128; wave: rt = wave&3, 8 col-tiles each,
  // next-tile B fragments prefetched ahead of the WMMA chain.
  {
    const int rt = wave & 3, ntb = (wave >> 2) * 8;
    const __bf16* Arow = sX + (rt * 16 + lrow) * LDK128;
    v16bf af[4], bf[4], bn[4];
#pragma unroll
    for (int kc = 0; kc < 4; ++kc) af[kc] = load_frag(Arow, kc * 32, kg);
    const __bf16* Brow0 = sW1 + (ntb * 16 + lrow) * LDK128;
#pragma unroll
    for (int kc = 0; kc < 4; ++kc) bf[kc] = load_frag(Brow0, kc * 32, kg);
#pragma unroll
    for (int t = 0; t < 8; ++t) {
      const int nt = ntb + t;
      if (t < 7) {
        const __bf16* Bn = sW1 + ((nt + 1) * 16 + lrow) * LDK128;
#pragma unroll
        for (int kc = 0; kc < 4; ++kc) bn[kc] = load_frag(Bn, kc * 32, kg);
      }
      v8f c = {};
#pragma unroll
      for (int kc = 0; kc < 4; ++kc) c = wmma_bf16(af[kc], bf[kc], c);
      const int ncol = nt * 16 + lrow;
      const float bv = b1[ncol];
      const int crow0 = rt * 16 + 8 * kg;
#pragma unroll
      for (int i = 0; i < 8; ++i) {
        float v = c[i] + bv;
        sH[(crow0 + i) * LDK256 + ncol] = (__bf16)(v > 0.f ? v : 0.f);
      }
      if (t < 7) {
#pragma unroll
        for (int kc = 0; kc < 4; ++kc) bf[kc] = bn[kc];
      }
    }
  }
  __syncthreads();

  // GEMM2: 64x16 over K=256; waves 0..3, 1-chunk lookahead on A and B.
  if (wave < 4) {
    const int rt = wave;
    const __bf16* Arow = sH + (rt * 16 + lrow) * LDK256;
    const __bf16* Brow = sW2 + lrow * LDK256;
    v16bf ac = load_frag(Arow, 0, kg);
    v16bf bc = load_frag(Brow, 0, kg);
    v8f c = {};
#pragma unroll
    for (int kc = 0; kc < 8; ++kc) {
      v16bf an = ac, bn2 = bc;
      if (kc < 7) {
        an  = load_frag(Arow, (kc + 1) * 32, kg);
        bn2 = load_frag(Brow, (kc + 1) * 32, kg);
      }
      c = wmma_bf16(ac, bc, c);
      ac = an; bc = bn2;
    }
    const float bv = b2[lrow];
    const int crow0 = row0 + rt * 16 + 8 * kg;
#pragma unroll
    for (int i = 0; i < 8; ++i) out[(crow0 + i) * NC + lrow] = c[i] + bv;
  }
}

// ---------------------------------------------------------------------------
extern "C" void kernel_launch(void* const* d_in, const int* in_sizes, int n_in,
                              void* d_out, int out_size, void* d_ws, size_t ws_size,
                              hipStream_t stream) {
  const int*   entity_ids = (const int*)d_in[0];
  const int*   module_ids = (const int*)d_in[1];
  const float* embedding  = (const float*)d_in[2];
  const float* W_in   = (const float*)d_in[3];
  const float* b_in   = (const float*)d_in[4];
  const float* W_bias = (const float*)d_in[5];
  const float* b_bias = (const float*)d_in[6];
  const float* W_f    = (const float*)d_in[7];
  const float* b_f    = (const float*)d_in[8];
  const float* W1     = (const float*)d_in[9];
  const float* b1     = (const float*)d_in[10];
  const float* W2     = (const float*)d_in[11];
  const float* b2     = (const float*)d_in[12];
  float* out = (float*)d_out;

  char* ws = (char*)d_ws;
  __bf16* Wint = (__bf16*)(ws + OFF_WIN);
  __bf16* Wbit = (__bf16*)(ws + OFF_WBI);
  __bf16* Wft  = (__bf16*)(ws + OFF_WF);
  __bf16* W1t  = (__bf16*)(ws + OFF_W1);
  __bf16* W2t  = (__bf16*)(ws + OFF_W2);
  int*    counts = (int*)(ws + OFF_CNT);
  int*    lists  = (int*)(ws + OFF_LST);
  float*  X0 = (float*)(ws + OFF_X0);
  float*  X1 = (float*)(ws + OFF_X1);

  // 131072 + 262144 + 32768 + 4096 = 430080 elements -> 1680 blocks of 256
  prep_kernel<<<1680, 256, 0, stream>>>(W_in, W_bias, W_f, W1, W2,
                                        Wint, Wbit, Wft, W1t, W2t);
  zero_counts<<<1, 32, 0, stream>>>(counts);
  bucketize<<<(NSTEPS * NB + 255) / 256, 256, 0, stream>>>(module_ids, counts, lists);
  gather_x0<<<(NB * NE) / 256, 256, 0, stream>>>(entity_ids, embedding, X0);

  dim3 sgrid(NB / ROWS, NM);
  step_kernel<<<sgrid, 256, STEP_LDS, stream>>>(X0, X1, embedding, entity_ids,
      b_in, b_bias, b_f, Wint, Wbit, Wft, counts, lists, 0);
  step_kernel<<<sgrid, 256, STEP_LDS, stream>>>(X1, X0, embedding, entity_ids,
      b_in, b_bias, b_f, Wint, Wbit, Wft, counts, lists, 1);
  step_kernel<<<sgrid, 256, STEP_LDS, stream>>>(X0, X1, embedding, entity_ids,
      b_in, b_bias, b_f, Wint, Wbit, Wft, counts, lists, 2);

  head_kernel<<<NB / ROWS, 256, HEAD_LDS, stream>>>(X1, W1t, W2t, b1, b2, out);
}
